// SimpleRoPEAttentionModel_10187662426526
// MI455X (gfx1250) — matrix operand order repaired and checked by reading, not compiled, not run on hardware
//
#include <hip/hip_runtime.h>

#define BB 2
#define SS 2048
#define EE 2048
#define HH 16
#define DD 128
#define MM (BB * SS) // 4096

typedef __bf16 bf16_t;
typedef __attribute__((ext_vector_type(16))) __bf16 v16bf;
typedef __attribute__((ext_vector_type(8)))  __bf16 v8bf;
typedef __attribute__((ext_vector_type(8)))  float  v8f;

static __device__ __forceinline__ bf16_t f2bf(float f) {
  unsigned u = __builtin_bit_cast(unsigned, f);
  unsigned r = u + 0x7FFFu + ((u >> 16) & 1u);   // round-to-nearest-even
  unsigned short h = (unsigned short)(r >> 16);
  return __builtin_bit_cast(bf16_t, h);
}
static __device__ __forceinline__ float bf2f(bf16_t b) {
  unsigned short h = __builtin_bit_cast(unsigned short, b);
  unsigned u = ((unsigned)h) << 16;
  return __builtin_bit_cast(float, u);
}
static __device__ __forceinline__ v16bf join16(v8bf lo, v8bf hi) {
  v16bf r;
#pragma unroll
  for (int e = 0; e < 8; ++e) { r[e] = lo[e]; r[e + 8] = hi[e]; }
  return r;
}
static __device__ __forceinline__ float rmax16(float v) {
#pragma unroll
  for (int m = 1; m < 16; m <<= 1) v = fmaxf(v, __shfl_xor(v, m, 32));
  return v;
}
static __device__ __forceinline__ float rsum16(float v) {
#pragma unroll
  for (int m = 1; m < 16; m <<= 1) v += __shfl_xor(v, m, 32);
  return v;
}

// ---- CDNA5 async global->LDS copy (ASYNCcnt path, see CDNA5 ISA) ----
static __device__ __forceinline__ unsigned lds_off(const void* p) {
  // generic pointer to LDS: low 32 bits are the LDS byte offset
  return (unsigned)(unsigned long long)p;
}
static __device__ __forceinline__ void async_ld_b128(unsigned lds, const void* g) {
  asm volatile("global_load_async_to_lds_b128 %0, %1, off"
               :: "v"(lds), "v"((unsigned long long)g)
               : "memory");
}
static __device__ __forceinline__ void wait_async0() {
  asm volatile("s_wait_asynccnt 0x0" ::: "memory");
}

// ---------------------------------------------------------------- convert
__global__ void cvt_f32_to_bf16(const float* __restrict__ src,
                                bf16_t* __restrict__ dst, int n) {
  int i = blockIdx.x * blockDim.x + threadIdx.x;
  int stride = gridDim.x * blockDim.x;
  for (; i < n; i += stride) dst[i] = f2bf(src[i]);
}

// ---------------------------------------------------------------- GEMM: A[M,K] @ W[N,K]^T
// Double-buffered LDS, async global->LDS pipeline, 8 WMMAs per stage per wave.
// mode 0: out bf16 scattered to [B,H,S,D]   (QKV projections)
// mode 1: out f32 at [M,N]=[B*S,E]          (final projection)
__global__ __launch_bounds__(256) void gemm_bf16_xwT(
    const bf16_t* __restrict__ A, const bf16_t* __restrict__ W,
    void* __restrict__ out, int mode) {
  __shared__ __align__(16) bf16_t lA[2][128][32];
  __shared__ __align__(16) bf16_t lB[2][128][32]; // [n][k]
  const int tid = threadIdx.x;
  const int m0 = blockIdx.y * 128;
  const int n0 = blockIdx.x * 128;
  const int w = tid >> 5, lane = tid & 31;
  const int wm = w & 3, wn = w >> 2;              // 4x2 wave grid
  const int laneRow = lane & 15, laneHi = lane >> 4;

  // per-thread copy coordinates (256 threads x 4 x b128 = two 128x32 tiles)
  const int lr = tid >> 2;          // 0..63
  const int lk = (tid & 3) * 8;     // 0,8,16,24

  // prologue: stage 0 via async copies
  async_ld_b128(lds_off(&lA[0][lr][lk]),      &A[(size_t)(m0 + lr)      * EE + lk]);
  async_ld_b128(lds_off(&lA[0][lr + 64][lk]), &A[(size_t)(m0 + lr + 64) * EE + lk]);
  async_ld_b128(lds_off(&lB[0][lr][lk]),      &W[(size_t)(n0 + lr)      * EE + lk]);
  async_ld_b128(lds_off(&lB[0][lr + 64][lk]), &W[(size_t)(n0 + lr + 64) * EE + lk]);
  wait_async0();
  __syncthreads();

  v8f acc[2][4] = {};

  int buf = 0;
  for (int k0 = 0; k0 < EE; k0 += 32, buf ^= 1) {
    const int kn = k0 + 32;
    if (kn < EE) { // stream next stage into the other buffer
      async_ld_b128(lds_off(&lA[buf ^ 1][lr][lk]),      &A[(size_t)(m0 + lr)      * EE + kn + lk]);
      async_ld_b128(lds_off(&lA[buf ^ 1][lr + 64][lk]), &A[(size_t)(m0 + lr + 64) * EE + kn + lk]);
      async_ld_b128(lds_off(&lB[buf ^ 1][lr][lk]),      &W[(size_t)(n0 + lr)      * EE + kn + lk]);
      async_ld_b128(lds_off(&lB[buf ^ 1][lr + 64][lk]), &W[(size_t)(n0 + lr + 64) * EE + kn + lk]);
    }

    v16bf af[2], bfr[4];
#pragma unroll
    for (int mi = 0; mi < 2; ++mi) {
      int row = wm * 32 + mi * 16 + laneRow;
      int kb = laneHi * 8;
      af[mi] = join16(*(const v8bf*)&lA[buf][row][kb],
                      *(const v8bf*)&lA[buf][row][16 + kb]);
    }
#pragma unroll
    for (int ni = 0; ni < 4; ++ni) {
      int col = wn * 64 + ni * 16 + laneRow;
      int kb = laneHi * 8;
      bfr[ni] = join16(*(const v8bf*)&lB[buf][col][kb],
                       *(const v8bf*)&lB[buf][col][kb + 16]);
    }
#pragma unroll
    for (int mi = 0; mi < 2; ++mi)
#pragma unroll
      for (int ni = 0; ni < 4; ++ni)
        acc[mi][ni] = __builtin_amdgcn_wmma_f32_16x16x32_bf16(
            false, af[mi], false, bfr[ni], (short)0, acc[mi][ni], false, false);

    wait_async0();   // our async copies for stage k+1 are in LDS
    __syncthreads(); // everyone done reading buf / writing buf^1
  }

#pragma unroll
  for (int mi = 0; mi < 2; ++mi)
#pragma unroll
    for (int ni = 0; ni < 4; ++ni)
#pragma unroll
      for (int r = 0; r < 8; ++r) {
        int mg = m0 + wm * 32 + mi * 16 + r + 8 * laneHi;
        int ng = n0 + wn * 64 + ni * 16 + laneRow;
        float v = acc[mi][ni][r];
        if (mode == 0) {
          int b = mg >> 11, s = mg & (SS - 1);
          int h = ng >> 7,  d = ng & (DD - 1);
          ((bf16_t*)out)[(((size_t)b * HH + h) * SS + s) * DD + d] = f2bf(v);
        } else {
          ((float*)out)[(size_t)mg * EE + ng] = v;
        }
      }
}

// ---------------------------------------------------------------- RoPE (NeoX rotate-half), in place on [B,H,S,D]
__global__ void rope_qk(bf16_t* __restrict__ Q, bf16_t* __restrict__ Kt,
                        const float* __restrict__ cosv,
                        const float* __restrict__ sinv) {
  const int half = DD / 2;
  int idx = blockIdx.x * blockDim.x + threadIdx.x;
  int total = BB * HH * SS * half;
  if (idx >= total) return;
  int i = idx % half;
  int s = (idx / half) % SS;
  int bh = idx / (half * SS);
  size_t base = ((size_t)bh * SS + s) * DD;
  float c0 = cosv[s * DD + i],        s0 = sinv[s * DD + i];
  float c1 = cosv[s * DD + i + half], s1 = sinv[s * DD + i + half];
  {
    float a = bf2f(Q[base + i]), b = bf2f(Q[base + i + half]);
    Q[base + i]        = f2bf(a * c0 - b * s0);
    Q[base + i + half] = f2bf(b * c1 + a * s1);
  }
  {
    float a = bf2f(Kt[base + i]), b = bf2f(Kt[base + i + half]);
    Kt[base + i]        = f2bf(a * c0 - b * s0);
    Kt[base + i + half] = f2bf(b * c1 + a * s1);
  }
}

// ---------------------------------------------------------------- causal flash attention
// grid: (S/64 q-tiles, B*H). block: 128 (4 waves, 16 q rows each).
__global__ __launch_bounds__(128) void flash_attn_causal(
    const bf16_t* __restrict__ Q, const bf16_t* __restrict__ K,
    const bf16_t* __restrict__ V, bf16_t* __restrict__ O) {
  __shared__ __align__(16) bf16_t lK[64][128];
  __shared__ __align__(16) bf16_t lVt[128][72];    // V transposed [d][k]
  __shared__ __align__(16) bf16_t lP[4][16][64];   // per-wave P staging
  const int bh = blockIdx.y;
  const int qt = blockIdx.x;
  const int tid = threadIdx.x;
  const int w = tid >> 5, lane = tid & 31;
  const int laneRow = lane & 15, laneHi = lane >> 4;
  const size_t base = (size_t)bh * SS * DD;
  const int q0 = qt * 64;
  const float scale = 0.08838834764831845f; // 1/sqrt(128), folded into Q

  v16bf qf[4];
  {
    int qrow = q0 + w * 16 + laneRow;
#pragma unroll
    for (int ks = 0; ks < 4; ++ks) {
      int kb = ks * 32 + laneHi * 8;
      v8bf lo = *(const v8bf*)&Q[base + (size_t)qrow * DD + kb];
      v8bf hi = *(const v8bf*)&Q[base + (size_t)qrow * DD + kb + 16];
#pragma unroll
      for (int e = 0; e < 8; ++e) {
        qf[ks][e]     = f2bf(bf2f(lo[e]) * scale);
        qf[ks][e + 8] = f2bf(bf2f(hi[e]) * scale);
      }
    }
  }

  v8f of[8] = {};
  float mrow[8], lrow[8];
#pragma unroll
  for (int r = 0; r < 8; ++r) { mrow[r] = -3.0e38f; lrow[r] = 0.0f; }

  for (int kt = 0; kt <= qt; ++kt) {
    __syncthreads();
#pragma unroll
    for (int it = 0; it < 8; ++it) {
      int chunk = it * 128 + tid;
      int r = chunk >> 4;
      int c = (chunk & 15) * 8;
      // K tile: direct async global->LDS (no VGPR bounce)
      async_ld_b128(lds_off(&lK[r][c]), &K[base + (size_t)(kt * 64 + r) * DD + c]);
      // V tile: via VGPRs, transposed into lVt
      v8bf vv = *(const v8bf*)&V[base + (size_t)(kt * 64 + r) * DD + c];
#pragma unroll
      for (int e = 0; e < 8; ++e) lVt[c + e][r] = vv[e];
    }
    wait_async0();
    __syncthreads();

    // S = Q @ K^T   (B-frag column = a K row, contiguous in d)
    v8f sacc[4] = {};
#pragma unroll
    for (int ni = 0; ni < 4; ++ni)
#pragma unroll
      for (int ks = 0; ks < 4; ++ks) {
        int col = ni * 16 + laneRow;
        int kb = ks * 32 + laneHi * 8;
        v16bf bfrag = join16(*(const v8bf*)&lK[col][kb],
                             *(const v8bf*)&lK[col][kb + 16]);
        sacc[ni] = __builtin_amdgcn_wmma_f32_16x16x32_bf16(
            false, qf[ks], false, bfrag, (short)0, sacc[ni], false, false);
      }

    // causal mask + online softmax (C layout: row = r + 8*laneHi, col = laneRow)
#pragma unroll
    for (int r = 0; r < 8; ++r) {
      int qg = q0 + w * 16 + r + 8 * laneHi;
      float t = -3.0e38f;
#pragma unroll
      for (int ni = 0; ni < 4; ++ni) {
        int kg = kt * 64 + ni * 16 + laneRow;
        float v = sacc[ni][r];
        if (kg > qg) v = -3.0e38f;
        sacc[ni][r] = v;
        t = fmaxf(t, v);
      }
      t = rmax16(t);
      float mnew = fmaxf(mrow[r], t);
      float alpha = __expf(mrow[r] - mnew);
      mrow[r] = mnew;
      lrow[r] *= alpha;
#pragma unroll
      for (int nd = 0; nd < 8; ++nd) of[nd][r] *= alpha;
      float sum = 0.0f;
#pragma unroll
      for (int ni = 0; ni < 4; ++ni) {
        float p = __expf(sacc[ni][r] - mnew);
        sum += p;
        lP[w][r + 8 * laneHi][ni * 16 + laneRow] = f2bf(p);
      }
      lrow[r] += rsum16(sum);
    }

    // O += P @ V  (P re-read from LDS in A layout; V from transposed LDS)
    v16bf pf[2];
#pragma unroll
    for (int ks = 0; ks < 2; ++ks) {
      int kb = ks * 32 + laneHi * 8;
      pf[ks] = join16(*(const v8bf*)&lP[w][laneRow][kb],
                      *(const v8bf*)&lP[w][laneRow][kb + 16]);
    }
#pragma unroll
    for (int nd = 0; nd < 8; ++nd)
#pragma unroll
      for (int ks = 0; ks < 2; ++ks) {
        int col = nd * 16 + laneRow;
        int kb = ks * 32 + laneHi * 8;
        v16bf vfrag = join16(*(const v8bf*)&lVt[col][kb],
                             *(const v8bf*)&lVt[col][kb + 16]);
        of[nd] = __builtin_amdgcn_wmma_f32_16x16x32_bf16(
            false, pf[ks], false, vfrag, (short)0, of[nd], false, false);
      }
  }

  // normalize + write [B,S,E] bf16
  const int b = bh / HH, h = bh % HH;
#pragma unroll
  for (int r = 0; r < 8; ++r) {
    int s = q0 + w * 16 + r + 8 * laneHi;
    float inv = 1.0f / lrow[r];
#pragma unroll
    for (int nd = 0; nd < 8; ++nd) {
      int d = nd * 16 + laneRow;
      O[((size_t)b * SS + s) * EE + h * DD + d] = f2bf(of[nd][r] * inv);
    }
  }
}

// ---------------------------------------------------------------- launcher
extern "C" void kernel_launch(void* const* d_in, const int* in_sizes, int n_in,
                              void* d_out, int out_size, void* d_ws, size_t ws_size,
                              hipStream_t stream) {
  (void)in_sizes; (void)n_in; (void)out_size; (void)ws_size;
  const float* x    = (const float*)d_in[0];
  const float* cosv = (const float*)d_in[1];
  const float* sinv = (const float*)d_in[2];
  const float* Wq   = (const float*)d_in[3];
  const float* Wk   = (const float*)d_in[4];
  const float* Wv   = (const float*)d_in[5];
  const float* Wo   = (const float*)d_in[6];

  char* ws = (char*)d_ws;
  size_t off = 0;
  bf16_t* xb  = (bf16_t*)(ws + off); off += (size_t)MM * EE * sizeof(bf16_t);
  bf16_t* wqb = (bf16_t*)(ws + off); off += (size_t)EE * EE * sizeof(bf16_t);
  bf16_t* wkb = (bf16_t*)(ws + off); off += (size_t)EE * EE * sizeof(bf16_t);
  bf16_t* wvb = (bf16_t*)(ws + off); off += (size_t)EE * EE * sizeof(bf16_t);
  bf16_t* wob = (bf16_t*)(ws + off); off += (size_t)EE * EE * sizeof(bf16_t);
  bf16_t* Qb  = (bf16_t*)(ws + off); off += (size_t)BB * HH * SS * DD * sizeof(bf16_t);
  bf16_t* Kb  = (bf16_t*)(ws + off); off += (size_t)BB * HH * SS * DD * sizeof(bf16_t);
  bf16_t* Vb  = (bf16_t*)(ws + off); off += (size_t)BB * HH * SS * DD * sizeof(bf16_t);
  bf16_t* Ob  = (bf16_t*)(ws + off); off += (size_t)MM * EE * sizeof(bf16_t);

  const int nX = MM * EE;
  const int nW = EE * EE;
  cvt_f32_to_bf16<<<4096, 256, 0, stream>>>(x,  xb,  nX);
  cvt_f32_to_bf16<<<2048, 256, 0, stream>>>(Wq, wqb, nW);
  cvt_f32_to_bf16<<<2048, 256, 0, stream>>>(Wk, wkb, nW);
  cvt_f32_to_bf16<<<2048, 256, 0, stream>>>(Wv, wvb, nW);
  cvt_f32_to_bf16<<<2048, 256, 0, stream>>>(Wo, wob, nW);

  dim3 gg(EE / 128, MM / 128); // (16, 32)
  gemm_bf16_xwT<<<gg, 256, 0, stream>>>(xb, wqb, Qb, 0);
  gemm_bf16_xwT<<<gg, 256, 0, stream>>>(xb, wkb, Kb, 0);
  gemm_bf16_xwT<<<gg, 256, 0, stream>>>(xb, wvb, Vb, 0);

  const int ropeN = BB * HH * SS * (DD / 2);
  rope_qk<<<(ropeN + 255) / 256, 256, 0, stream>>>(Qb, Kb, cosv, sinv);

  flash_attn_causal<<<dim3(SS / 64, BB * HH), 128, 0, stream>>>(Qb, Kb, Vb, Ob);

  gemm_bf16_xwT<<<gg, 256, 0, stream>>>(Ob, wob, d_out, 1);
}
